// VQEmbedding_84052509982993
// MI455X (gfx1250) — compile-verified, matrix operand order checked by reading, not run on hardware
//
#include <hip/hip_runtime.h>
#include <hip/hip_bf16.h>

// ---------------------------------------------------------------------------
// VQ-VAE vector quantization for MI455X (gfx1250, wave32, WMMA).
//   z:        [16,2048,512] f32  -> N=32768 rows, D=512
//   codebook: [1024,512]    f32  -> K=1024
// distances argmin via  ||e_k||^2 - 2 z.e_k  with the GEMM done as
// split-bf16 (hi+lo) WMMA:  dot ~= hi*hi + hi*lo + lo*hi  (3 wmma / 32-chunk)
// ---------------------------------------------------------------------------

typedef __attribute__((ext_vector_type(16))) __bf16 v16bf;
typedef __attribute__((ext_vector_type(8)))  __bf16 v8bf;
typedef __attribute__((ext_vector_type(8)))  float  v8f;

namespace vq {
constexpr int D    = 512;
constexpr int K    = 1024;
constexpr int N    = 32768;
constexpr int RPB  = 128;              // rows of z per block
constexpr int NBLK = N / RPB;          // 256 blocks
constexpr size_t ZQ_ELEMS = (size_t)N * D;      // 16777216
constexpr size_t LOSS_OFF = ZQ_ELEMS;           // 1 float
constexpr size_t IDX_OFF  = ZQ_ELEMS + 1;       // 32768 floats
}

// Build a 16x32 bf16 WMMA fragment (A layout; B of A*B^T uses the identical
// pattern).  Lane l holds matrix row (l&15); group g = l>>4 selects the
// d-subchunks [8g,8g+8) and [16+8g,16+8g+8).  Two 16-byte loads per fragment.
__device__ __forceinline__ v16bf vq_load_frag(const __bf16* __restrict__ row,
                                              int d0, int g) {
  v8bf x0 = *(const v8bf*)(row + d0 + 8 * g);
  v8bf x1 = *(const v8bf*)(row + d0 + 16 + 8 * g);
  v16bf r;
#pragma unroll
  for (int i = 0; i < 8; ++i) { r[i] = x0[i]; r[8 + i] = x1[i]; }
  return r;
}

// ---------------------------------------------------------------------------
// Prep: codebook f32 -> bf16 hi/lo planes (L2-resident, 2MB) + exact norms.
// One block per codebook row, 128 threads (4 waves).
// ---------------------------------------------------------------------------
__global__ __launch_bounds__(128) void vq_prep_kernel(
    const float* __restrict__ cb, __bf16* __restrict__ cbhi,
    __bf16* __restrict__ cblo, float* __restrict__ cbnorm) {
  const int k   = blockIdx.x;
  const int tid = threadIdx.x;
  const float4* row = (const float4*)(cb + (size_t)k * vq::D);
  float4 v = row[tid];                      // 128 threads x float4 = 512 elems
  float s = 0.f;
  float xs[4] = {v.x, v.y, v.z, v.w};
#pragma unroll
  for (int i = 0; i < 4; ++i) {
    float x  = xs[i];
    __bf16 h = (__bf16)x;
    __bf16 l = (__bf16)(x - (float)h);
    cbhi[(size_t)k * vq::D + tid * 4 + i] = h;
    cblo[(size_t)k * vq::D + tid * 4 + i] = l;
    s += x * x;
  }
#pragma unroll
  for (int m = 16; m > 0; m >>= 1) s += __shfl_xor(s, m, 32);
  __shared__ float red[4];
  const int lane = tid & 31, wave = tid >> 5;
  if (lane == 0) red[wave] = s;
  __syncthreads();
  if (tid == 0) cbnorm[k] = red[0] + red[1] + red[2] + red[3];
}

// ---------------------------------------------------------------------------
// Main: per-block 128 rows.  Phase 1: f32->bf16 hi/lo split into LDS (once).
// Phase 2: per-wave 16x16 WMMA tiles over all 64 k-tiles, running argmin.
// Phase 3: gather z_q from the f32 codebook, write outputs, loss partial.
// ---------------------------------------------------------------------------
__global__ __launch_bounds__(256) void vq_main_kernel(
    const float* __restrict__ z, const float* __restrict__ cb,
    const __bf16* __restrict__ cbhi, const __bf16* __restrict__ cblo,
    const float* __restrict__ cbnorm, float* __restrict__ out,
    float* __restrict__ partials) {
  __shared__ __align__(16) __bf16 sHi[vq::RPB][vq::D];   // 128 KB
  __shared__ __align__(16) __bf16 sLo[vq::RPB][vq::D];   // 128 KB
  __shared__ int   sIdx[vq::RPB];
  __shared__ float sRed[8];

  const int tid      = threadIdx.x;
  const int blockRow = blockIdx.x * vq::RPB;
  const float4* zb   = (const float4*)(z + (size_t)blockRow * vq::D);

  // ---- Phase 1: split z tile into bf16 hi/lo planes in LDS -------------
  for (int e = tid; e < vq::RPB * (vq::D / 4); e += 256) {
    float4 v = zb[e];
    const int r = e >> 7;                 // / (D/4)
    const int c = (e & 127) << 2;
    float xs[4] = {v.x, v.y, v.z, v.w};
#pragma unroll
    for (int i = 0; i < 4; ++i) {
      float x  = xs[i];
      __bf16 h = (__bf16)x;
      __bf16 l = (__bf16)(x - (float)h);
      sHi[r][c + i] = h;
      sLo[r][c + i] = l;
    }
  }
  __syncthreads();

  // ---- Phase 2: WMMA distance sweep + running argmin -------------------
  const int lane = tid & 31;
  const int wave = tid >> 5;
  const int g    = lane >> 4;             // fragment half
  const int m    = lane & 15;             // row (A) / column (B) within tile
  const __bf16* aHiRow = &sHi[wave * 16 + m][0];
  const __bf16* aLoRow = &sLo[wave * 16 + m][0];

  float best[8];
  int   bidx[8];
#pragma unroll
  for (int j = 0; j < 8; ++j) { best[j] = 3.4e38f; bidx[j] = 0; }

  for (int kt = 0; kt < vq::K / 16; ++kt) {
    const int k = kt * 16 + m;            // this lane's codebook row
    const __bf16* bHiRow = cbhi + (size_t)k * vq::D;
    const __bf16* bLoRow = cblo + (size_t)k * vq::D;
    if (kt + 1 < vq::K / 16) {            // warm next k-tile (global_prefetch)
      __builtin_prefetch(bHiRow + (size_t)16 * vq::D, 0, 1);
      __builtin_prefetch(bLoRow + (size_t)16 * vq::D, 0, 1);
    }
    v8f acc = {};
#pragma unroll 4
    for (int dc = 0; dc < vq::D / 32; ++dc) {
      const int d0 = dc * 32;
      v16bf ahi = vq_load_frag(aHiRow, d0, g);
      v16bf alo = vq_load_frag(aLoRow, d0, g);
      v16bf bhi = vq_load_frag(bHiRow, d0, g);
      v16bf blo = vq_load_frag(bLoRow, d0, g);
      acc = __builtin_amdgcn_wmma_f32_16x16x32_bf16(false, ahi, false, bhi,
                                                    (short)0, acc, false, false);
      acc = __builtin_amdgcn_wmma_f32_16x16x32_bf16(false, ahi, false, blo,
                                                    (short)0, acc, false, false);
      acc = __builtin_amdgcn_wmma_f32_16x16x32_bf16(false, alo, false, bhi,
                                                    (short)0, acc, false, false);
    }
    const float nrm = cbnorm[k];          // dist(+const) = ||e||^2 - 2 z.e
#pragma unroll
    for (int j = 0; j < 8; ++j) {
      const float dist = fmaf(-2.0f, acc[j], nrm);
      if (dist < best[j]) { best[j] = dist; bidx[j] = k; }   // first-k wins ties
    }
  }

  // argmin across the 16 column lanes (pairwise, lower index breaks ties)
#pragma unroll
  for (int s = 1; s < 16; s <<= 1) {
#pragma unroll
    for (int j = 0; j < 8; ++j) {
      const float ov = __shfl_xor(best[j], s, 32);
      const int   oi = __shfl_xor(bidx[j], s, 32);
      if (ov < best[j] || (ov == best[j] && oi < bidx[j])) {
        best[j] = ov; bidx[j] = oi;
      }
    }
  }
  if (m == 0) {                           // lanes 0 and 16: rows 8g+j
#pragma unroll
    for (int j = 0; j < 8; ++j) sIdx[wave * 16 + 8 * g + j] = bidx[j];
  }
  __syncthreads();

  // ---- Phase 3: outputs ------------------------------------------------
  if (tid < vq::RPB)
    out[vq::IDX_OFF + blockRow + tid] = (float)sIdx[tid];

  float sum = 0.f;
  for (int e = tid; e < vq::RPB * (vq::D / 4); e += 256) {
    const int r  = e >> 7;
    const int c4 = e & 127;
    const float4 q  = ((const float4*)(cb + (size_t)sIdx[r] * vq::D))[c4];
    const float4 zv = zb[e];
    ((float4*)out)[(size_t)blockRow * (vq::D / 4) + e] = q;  // z_q_st == z_q
    const float dx = q.x - zv.x, dy = q.y - zv.y;
    const float dz = q.z - zv.z, dw = q.w - zv.w;
    sum += dx * dx + dy * dy + dz * dz + dw * dw;
  }
#pragma unroll
  for (int s = 16; s > 0; s >>= 1) sum += __shfl_xor(sum, s, 32);
  if (lane == 0) sRed[wave] = sum;
  __syncthreads();
  if (tid == 0) {
    float t = 0.f;
#pragma unroll
    for (int i = 0; i < 8; ++i) t += sRed[i];
    partials[blockIdx.x] = t;             // deterministic (no float atomics)
  }
}

// ---------------------------------------------------------------------------
// Finish: reduce 256 per-block partials -> loss = 1.25 * sum / numel
// ---------------------------------------------------------------------------
__global__ __launch_bounds__(256) void vq_finish_kernel(
    const float* __restrict__ partials, float* __restrict__ out) {
  __shared__ float red[8];
  const int tid = threadIdx.x;
  float v = partials[tid];
#pragma unroll
  for (int s = 16; s > 0; s >>= 1) v += __shfl_xor(v, s, 32);
  const int lane = tid & 31, wave = tid >> 5;
  if (lane == 0) red[wave] = v;
  __syncthreads();
  if (tid == 0) {
    float t = 0.f;
#pragma unroll
    for (int i = 0; i < 8; ++i) t += red[i];
    out[vq::LOSS_OFF] = t * 1.25f / (float)vq::ZQ_ELEMS;
  }
}

// ---------------------------------------------------------------------------
extern "C" void kernel_launch(void* const* d_in, const int* in_sizes, int n_in,
                              void* d_out, int out_size, void* d_ws,
                              size_t ws_size, hipStream_t stream) {
  const float* z  = (const float*)d_in[0];
  const float* cb = (const float*)d_in[1];
  float* out      = (float*)d_out;

  char* ws = (char*)d_ws;
  __bf16* cbhi   = (__bf16*)ws;                                   // 1 MB
  __bf16* cblo   = (__bf16*)(ws + (size_t)vq::K * vq::D * 2);     // 1 MB
  float* cbnorm  = (float*)(ws + (size_t)vq::K * vq::D * 4);      // 4 KB
  float* partial = (float*)(ws + (size_t)vq::K * vq::D * 4 + vq::K * 4);

  vq_prep_kernel<<<vq::K, 128, 0, stream>>>(cb, cbhi, cblo, cbnorm);
  vq_main_kernel<<<vq::NBLK, 256, 0, stream>>>(z, cb, cbhi, cblo, cbnorm, out,
                                               partial);
  vq_finish_kernel<<<1, 256, 0, stream>>>(partial, out);
}
